// MambaMAE_21620865368264
// MI455X (gfx1250) — matrix operand order repaired and checked by reading, not compiled, not run on hardware
//
#include <hip/hip_runtime.h>
#include <cstdint>
#include <cstddef>

// ---------------- problem constants ----------------
#define BN    16
#define CINN  2
#define TT    4096
#define DMM   128
#define DII   256
#define DSZ   16
#define DRR   8
#define DCC   4
#define NLL   6
#define LSEQ  512                 // TT/8
#define MROWS (BN*LSEQ)           // 8192
#define NXZ   (2*DII)             // 512
#define NXP   48                  // xp output padded from 40 -> 48

typedef __attribute__((ext_vector_type(16))) _Float16 v16h;
typedef __attribute__((ext_vector_type(8)))  float    v8f;
typedef __attribute__((ext_vector_type(4)))  unsigned int u32x4;
typedef __attribute__((ext_vector_type(4)))  int i32x4;

#if __has_builtin(__builtin_amdgcn_global_load_async_to_lds_b128)
#define USE_ASYNC_LDS 1
#define GAS __attribute__((address_space(1)))
#define LAS __attribute__((address_space(3)))
#else
#define USE_ASYNC_LDS 0
#endif

// ---------------- helpers ----------------
__device__ __forceinline__ float gelu_exact(float v) {
  return 0.5f * v * (1.f + erff(v * 0.70710678118654752440f));
}
__device__ __forceinline__ float silu(float v) {
  return v / (1.f + __expf(-v));
}

// ---------------- WMMA GEMM: D = (Cres?) + A(MxK,f16) * B(NxK,f16)^T -> f32 ----------------
// Block = 8 waves sharing one N-panel of 16*NT columns. Each wave owns one 16-row M tile.
// B panel (contiguous in memory, row-major (N,K)) is staged into LDS via async
// global->LDS b128 loads when available; fragments are two b128 loads per lane, 16 halves apart.
template<int NT, int KDIM>
__global__ __launch_bounds__(256) void wmma_gemm(
    const _Float16* __restrict__ Ah, const _Float16* __restrict__ Bh,
    const float* __restrict__ Cres, float* __restrict__ Dout,
    int Mdim, int Ndim)
{
  const int wv   = threadIdx.x >> 5;
  const int lane = threadIdx.x & 31;
  const int hi   = lane >> 4;      // half-wave select
  const int mr   = lane & 15;
  const int m0   = (blockIdx.x * 8 + wv) * 16;
  const int n0   = blockIdx.y * (16 * NT);

#if USE_ASYNC_LDS
  __shared__ __align__(16) _Float16 Bl[NT * 16 * KDIM];
  {
    const int bytes = NT * 16 * KDIM * 2;
    const char* gsrc = (const char*)(Bh + (size_t)n0 * KDIM);
    for (int off = threadIdx.x * 16; off < bytes; off += 256 * 16) {
      __builtin_amdgcn_global_load_async_to_lds_b128(
          (GAS i32x4*)(uintptr_t)(gsrc + off),
          (LAS i32x4*)(uintptr_t)((char*)Bl + off),
          0, 0);
    }
    asm volatile("s_wait_asynccnt 0x0" ::: "memory");
    __syncthreads();
  }
  const _Float16* Bbase = Bl;                       // local col t*16+mr
#else
  const _Float16* Bbase = Bh + (size_t)n0 * KDIM;   // direct global
#endif

  union AF { v16h v; u32x4 q[2]; };

  v8f acc[NT];
  #pragma unroll
  for (int t = 0; t < NT; ++t) {
    #pragma unroll
    for (int r = 0; r < 8; ++r) {
      // C/D layout: VGPR r holds row m = r + 8*hi, col = lane&15
      acc[t][r] = Cres ? Cres[(size_t)(m0 + r + 8*hi) * Ndim + (n0 + t*16 + mr)] : 0.f;
    }
  }

  const _Float16* Arow = Ah + (size_t)(m0 + mr) * KDIM;
  __builtin_prefetch(Arow, 0, 1);

  #pragma unroll
  for (int k0 = 0; k0 < KDIM; k0 += 32) {
    // A frag: lanes0-15 halves [k0+0,8)+[k0+16,8); lanes16-31 +8  -> 2x b128
    AF af;
    af.q[0] = *(const u32x4*)(Arow + k0 + 8*hi);
    af.q[1] = *(const u32x4*)(Arow + k0 + 16 + 8*hi);
    #pragma unroll
    for (int t = 0; t < NT; ++t) {
      // B frag: lane n holds halves [k0+8*hi,8)+[k0+16+8*hi,8) of col n  -> 2x b128
      AF bf;
      const _Float16* bcol = Bbase + (size_t)(t*16 + mr) * KDIM + k0 + 8*hi;
      bf.q[0] = ((const u32x4*)bcol)[0];
      bf.q[1] = ((const u32x4*)bcol)[2];
      acc[t] = __builtin_amdgcn_wmma_f32_16x16x32_f16(
                   false, af.v, false, bf.v, (short)0, acc[t], false, false);
    }
  }

  #pragma unroll
  for (int t = 0; t < NT; ++t)
    #pragma unroll
    for (int r = 0; r < 8; ++r)
      Dout[(size_t)(m0 + r + 8*hi) * Ndim + (n0 + t*16 + mr)] = acc[t][r];
}

// ---------------- f32 -> f16 weight conversion (with row padding) ----------------
__global__ __launch_bounds__(256) void cvt_to_h(const float* __restrict__ src,
                                                _Float16* __restrict__ dst,
                                                int R, int C, int Rp)
{
  int idx = blockIdx.x * blockDim.x + threadIdx.x;
  if (idx >= Rp * C) return;
  int r = idx / C;
  dst[idx] = (r < R) ? (_Float16)src[idx] : (_Float16)0.f;
}

// ---------------- elementwise mask multiply ----------------
__global__ __launch_bounds__(256) void premask(const float* __restrict__ x,
                                               const float* __restrict__ m,
                                               float* __restrict__ out, int n)
{
  int i = blockIdx.x * blockDim.x + threadIdx.x;
  if (i < n) out[i] = x[i] * m[i];
}

// ---------------- conv1d stride-2 pad-1 k=3 ----------------
__global__ __launch_bounds__(256) void conv1d_s2(
    const float* __restrict__ in, const float* __restrict__ W,
    const float* __restrict__ bias, float* __restrict__ out,
    int Cin, int Cout, int Tin, int Tout)
{
  int idx = blockIdx.x * blockDim.x + threadIdx.x;
  int total = BN * Cout * Tout;
  if (idx >= total) return;
  int t  = idx % Tout;
  int co = (idx / Tout) % Cout;
  int b  = idx / (Tout * Cout);
  float acc = bias[co];
  for (int ci = 0; ci < Cin; ++ci) {
    #pragma unroll
    for (int k = 0; k < 3; ++k) {
      int ti = 2*t + k - 1;
      if (ti >= 0 && ti < Tin)
        acc += W[(co*Cin + ci)*3 + k] * in[((size_t)b*Cin + ci)*Tin + ti];
    }
  }
  out[idx] = acc;
}

// ---------------- transposed conv1d stride-2 pad-1 outpad-1 k=3 ----------------
// y[b,o,t] = bias[o] + sum_{i,k} W[i,o,2-k] * x[b,i,(t+k-1)/2]  (valid/even only)
__global__ __launch_bounds__(256) void deconv1d(
    const float* __restrict__ in, const float* __restrict__ W,
    const float* __restrict__ bias, float* __restrict__ out,
    int Cin, int Cout, int Tin, int Tout)
{
  int idx = blockIdx.x * blockDim.x + threadIdx.x;
  int total = BN * Cout * Tout;
  if (idx >= total) return;
  int t = idx % Tout;
  int o = (idx / Tout) % Cout;
  int b = idx / (Tout * Cout);
  float acc = bias[o];
  for (int i = 0; i < Cin; ++i) {
    #pragma unroll
    for (int k = 0; k < 3; ++k) {
      int p = t + k - 1;
      if (p >= 0 && (p & 1) == 0) {
        int src = p >> 1;
        if (src < Tin)
          acc += W[((size_t)i*Cout + o)*3 + (2-k)] * in[((size_t)b*Cin + i)*Tin + src];
      }
    }
  }
  out[idx] = acc;
}

// ---------------- GroupNorm(1 group) stats per batch ----------------
__global__ __launch_bounds__(256) void gn_stats(const float* __restrict__ x,
                                                float* __restrict__ mean,
                                                float* __restrict__ rstd, int npb)
{
  __shared__ float ssum[256];
  __shared__ float ssq[256];
  int b = blockIdx.x;
  const float* p = x + (size_t)b * npb;
  float s = 0.f, q = 0.f;
  for (int i = threadIdx.x; i < npb; i += 256) { float v = p[i]; s += v; q += v*v; }
  ssum[threadIdx.x] = s; ssq[threadIdx.x] = q;
  __syncthreads();
  for (int st = 128; st > 0; st >>= 1) {
    if ((int)threadIdx.x < st) { ssum[threadIdx.x] += ssum[threadIdx.x+st]; ssq[threadIdx.x] += ssq[threadIdx.x+st]; }
    __syncthreads();
  }
  if (threadIdx.x == 0) {
    float m = ssum[0] / npb;
    float v = ssq[0] / npb - m*m;
    mean[b] = m;
    rstd[b] = rsqrtf(v + 1e-5f);
  }
}

__global__ __launch_bounds__(256) void gn_apply(
    const float* __restrict__ x, const float* __restrict__ mean,
    const float* __restrict__ rstd, const float* __restrict__ g,
    const float* __restrict__ be, float* __restrict__ out,
    int C, int Tl, int do_gelu)
{
  int idx = blockIdx.x * blockDim.x + threadIdx.x;
  int total = BN * C * Tl;
  if (idx >= total) return;
  int c = (idx / Tl) % C;
  int b = idx / (Tl * C);
  float v = (x[idx] - mean[b]) * rstd[b] * g[c] + be[c];
  out[idx] = do_gelu ? gelu_exact(v) : v;
}

// ---------------- (B,DM,L) -> (B,L,DM) ----------------
__global__ __launch_bounds__(256) void nch_to_nhc(const float* __restrict__ in,
                                                  float* __restrict__ out)
{
  int idx = blockIdx.x * blockDim.x + threadIdx.x;
  if (idx >= BN*DMM*LSEQ) return;
  int l = idx % LSEQ;
  int d = (idx / LSEQ) % DMM;
  int b = idx / (LSEQ * DMM);
  out[((size_t)b*LSEQ + l)*DMM + d] = in[idx];
}

// ---------------- LayerNorm over DM=128 (one wave per row) ----------------
// mode 0: write f16 row-major (GEMM A operand). mode 1: write f32 transposed (B,DM,L).
__global__ __launch_bounds__(256) void ln_rows(
    const float* __restrict__ x, const float* __restrict__ g, const float* __restrict__ be,
    _Float16* __restrict__ dst_h, float* __restrict__ dst_t, int mode)
{
  int row  = blockIdx.x * 8 + (threadIdx.x >> 5);
  int lane = threadIdx.x & 31;
  const float* p = x + (size_t)row * DMM;
  float v[4];
  #pragma unroll
  for (int j = 0; j < 4; ++j) v[j] = p[lane + 32*j];
  float s = v[0]+v[1]+v[2]+v[3];
  float q = v[0]*v[0]+v[1]*v[1]+v[2]*v[2]+v[3]*v[3];
  #pragma unroll
  for (int m = 1; m < 32; m <<= 1) { s += __shfl_xor(s, m, 32); q += __shfl_xor(q, m, 32); }
  float mean = s * (1.f / DMM);
  float rstd = rsqrtf(q * (1.f / DMM) - mean*mean + 1e-5f);
  #pragma unroll
  for (int j = 0; j < 4; ++j) {
    int c = lane + 32*j;
    float o = (v[j] - mean) * rstd * g[c] + be[c];
    if (mode == 0) {
      dst_h[(size_t)row*DMM + c] = (_Float16)o;
    } else {
      int b = row / LSEQ, l = row % LSEQ;
      dst_t[((size_t)b*DMM + c)*LSEQ + l] = o;
    }
  }
}

// ---------------- causal depthwise conv (DC=4) + SiLU on xi = xz[:, :DI] ----------------
__global__ __launch_bounds__(256) void dwconv_silu(
    const float* __restrict__ xz, const float* __restrict__ cW,
    const float* __restrict__ cb, float* __restrict__ xc,
    _Float16* __restrict__ xch)
{
  int idx = blockIdx.x * blockDim.x + threadIdx.x;
  if (idx >= MROWS * DII) return;
  int d  = idx % DII;
  int m  = idx / DII;
  int l  = m % LSEQ;
  int mb = m - l;                      // b*LSEQ
  float acc = cb[d];
  #pragma unroll
  for (int k = 0; k < DCC; ++k) {
    int ls = l + k - (DCC - 1);
    if (ls >= 0) acc += cW[d*DCC + k] * xz[(size_t)(mb + ls)*NXZ + d];
  }
  float v = silu(acc);
  xc[idx]  = v;
  xch[idx] = (_Float16)v;
}

// ---------------- dt = softplus(dbl[:, :DR] @ dt_W^T + dt_b) ----------------
__global__ __launch_bounds__(256) void dt_softplus(
    const float* __restrict__ dbl, const float* __restrict__ dtW,
    const float* __restrict__ dtb, float* __restrict__ dt)
{
  int idx = blockIdx.x * blockDim.x + threadIdx.x;
  if (idx >= MROWS * DII) return;
  int d = idx % DII;
  int m = idx / DII;
  float a = dtb[d];
  #pragma unroll
  for (int r = 0; r < DRR; ++r) a += dbl[(size_t)m*NXP + r] * dtW[d*DRR + r];
  dt[idx] = (a > 20.f) ? a : log1pf(__expf(a));
}

// ---------------- selective scan: 16 lanes per (b,d) channel, sequential over L ----------------
__global__ __launch_bounds__(256) void ssm_scan(
    const float* __restrict__ dt, const float* __restrict__ xc,
    const float* __restrict__ dbl, const float* __restrict__ xz,
    const float* __restrict__ Alog, const float* __restrict__ Dp,
    _Float16* __restrict__ yh)
{
  int grp = blockIdx.x * 16 + (threadIdx.x >> 4);
  int si  = threadIdx.x & 15;
  if (grp >= BN * DII) return;
  int b = grp / DII, d = grp % DII;
  float Av = -__expf(Alog[d*DSZ + si]);
  float Dv = Dp[d];
  float h = 0.f;
  for (int l = 0; l < LSEQ; ++l) {
    int m = b*LSEQ + l;
    float dtv = dt[(size_t)m*DII + d];
    float xv  = xc[(size_t)m*DII + d];
    float Bv  = dbl[(size_t)m*NXP + DRR + si];
    float Cv  = dbl[(size_t)m*NXP + DRR + DSZ + si];
    h = h * __expf(dtv * Av) + dtv * Bv * xv;
    float yv = h * Cv;
    yv += __shfl_xor(yv, 1, 32);
    yv += __shfl_xor(yv, 2, 32);
    yv += __shfl_xor(yv, 4, 32);
    yv += __shfl_xor(yv, 8, 32);
    if (si == 0) {
      float zv = xz[(size_t)m*NXZ + DII + d];
      float o  = (yv + Dv * xv) * silu(zv);
      yh[(size_t)m*DII + d] = (_Float16)o;
    }
  }
}

// ---------------- copy x, mask into output tuple slots ----------------
__global__ __launch_bounds__(256) void tail_copy(const float* __restrict__ x,
                                                 const float* __restrict__ m,
                                                 float* __restrict__ out, int n)
{
  int i = blockIdx.x * blockDim.x + threadIdx.x;
  if (i < n) { out[n + i] = x[i]; out[2*n + i] = m[i]; }
}

// =====================================================================
extern "C" void kernel_launch(void* const* d_in, const int* in_sizes, int n_in,
                              void* d_out, int out_size, void* d_ws, size_t ws_size,
                              hipStream_t stream)
{
  (void)in_sizes; (void)n_in; (void)out_size; (void)ws_size;

  const float* x     = (const float*)d_in[0];
  const float* mask  = (const float*)d_in[1];
  const float* eW0   = (const float*)d_in[2];
  const float* eb0   = (const float*)d_in[3];
  const float* eg0   = (const float*)d_in[4];
  const float* ebe0  = (const float*)d_in[5];
  const float* eW1   = (const float*)d_in[6];
  const float* eb1   = (const float*)d_in[7];
  const float* eg1   = (const float*)d_in[8];
  const float* ebe1  = (const float*)d_in[9];
  const float* eW2   = (const float*)d_in[10];
  const float* eb2   = (const float*)d_in[11];
  const float* eg2   = (const float*)d_in[12];
  const float* ebe2  = (const float*)d_in[13];
  const float* ln_g  = (const float*)d_in[14];
  const float* ln_b  = (const float*)d_in[15];
  const float* in_W  = (const float*)d_in[16];
  const float* c_W   = (const float*)d_in[17];
  const float* c_b   = (const float*)d_in[18];
  const float* xp_W  = (const float*)d_in[19];
  const float* dt_W  = (const float*)d_in[20];
  const float* dt_b  = (const float*)d_in[21];
  const float* A_log = (const float*)d_in[22];
  const float* D_p   = (const float*)d_in[23];
  const float* out_W = (const float*)d_in[24];
  const float* nf_g  = (const float*)d_in[25];
  const float* nf_b  = (const float*)d_in[26];
  const float* dW0   = (const float*)d_in[27];
  const float* db0   = (const float*)d_in[28];
  const float* dg0   = (const float*)d_in[29];
  const float* dbe0  = (const float*)d_in[30];
  const float* dW1   = (const float*)d_in[31];
  const float* db1   = (const float*)d_in[32];
  const float* dg1   = (const float*)d_in[33];
  const float* dbe1  = (const float*)d_in[34];
  const float* dW2   = (const float*)d_in[35];
  const float* db2   = (const float*)d_in[36];

  // -------- workspace carve-out --------
  char* wp = (char*)d_ws;
  auto take = [&](size_t bytes) -> char* {
    char* r = wp; wp += (bytes + 255) & ~(size_t)255; return r;
  };
  float* hx    = (float*)take(sizeof(float) * BN*CINN*TT);
  float* e0    = (float*)take(sizeof(float) * BN*32*2048);
  float* e1    = (float*)take(sizeof(float) * BN*64*1024);
  float* e2    = (float*)take(sizeof(float) * BN*DMM*LSEQ);
  float* sbuf  = (float*)take(sizeof(float) * MROWS*DMM);
  float* xz    = (float*)take(sizeof(float) * (size_t)MROWS*NXZ);
  float* xc    = (float*)take(sizeof(float) * (size_t)MROWS*DII);
  float* dbl   = (float*)take(sizeof(float) * (size_t)MROWS*NXP);
  float* dtv   = (float*)take(sizeof(float) * (size_t)MROWS*DII);
  float* decin = (float*)take(sizeof(float) * BN*DMM*LSEQ);
  float* dd0   = (float*)take(sizeof(float) * BN*64*1024);
  float* dd1   = (float*)take(sizeof(float) * BN*32*2048);
  float* mean  = (float*)take(sizeof(float) * BN);
  float* rstd  = (float*)take(sizeof(float) * BN);
  _Float16* ulnh = (_Float16*)take(sizeof(_Float16) * (size_t)MROWS*DMM);
  _Float16* xch  = (_Float16*)take(sizeof(_Float16) * (size_t)MROWS*DII);
  _Float16* yh   = (_Float16*)take(sizeof(_Float16) * (size_t)MROWS*DII);
  _Float16* wih  = (_Float16*)take(sizeof(_Float16) * NLL*NXZ*DMM);
  _Float16* wxph = (_Float16*)take(sizeof(_Float16) * NLL*NXP*DII);
  _Float16* woh  = (_Float16*)take(sizeof(_Float16) * NLL*DMM*DII);

  dim3 blk(256);
  auto g1 = [](int n) { return dim3((unsigned)((n + 255) / 256)); };
  const int MB = MROWS / 128;     // 64 blocks of 8 m-tiles each

  // -------- weight f16 conversion (xp padded 40->48 rows/layer) --------
  cvt_to_h<<<g1(NLL*NXZ*DMM), blk, 0, stream>>>(in_W, wih, NLL*NXZ, DMM, NLL*NXZ);
  for (int i = 0; i < NLL; ++i)
    cvt_to_h<<<g1(NXP*DII), blk, 0, stream>>>(xp_W + (size_t)i*40*DII,
                                              wxph + (size_t)i*NXP*DII, 40, DII, NXP);
  cvt_to_h<<<g1(NLL*DMM*DII), blk, 0, stream>>>(out_W, woh, NLL*DMM, DII, NLL*DMM);

  // -------- encoder --------
  const int NXT = BN*CINN*TT;
  premask<<<g1(NXT), blk, 0, stream>>>(x, mask, hx, NXT);

  conv1d_s2<<<g1(BN*32*2048), blk, 0, stream>>>(hx, eW0, eb0, e0, CINN, 32, TT, 2048);
  gn_stats<<<BN, blk, 0, stream>>>(e0, mean, rstd, 32*2048);
  gn_apply<<<g1(BN*32*2048), blk, 0, stream>>>(e0, mean, rstd, eg0, ebe0, e0, 32, 2048, 1);

  conv1d_s2<<<g1(BN*64*1024), blk, 0, stream>>>(e0, eW1, eb1, e1, 32, 64, 2048, 1024);
  gn_stats<<<BN, blk, 0, stream>>>(e1, mean, rstd, 64*1024);
  gn_apply<<<g1(BN*64*1024), blk, 0, stream>>>(e1, mean, rstd, eg1, ebe1, e1, 64, 1024, 1);

  conv1d_s2<<<g1(BN*DMM*LSEQ), blk, 0, stream>>>(e1, eW2, eb2, e2, 64, DMM, 1024, LSEQ);
  gn_stats<<<BN, blk, 0, stream>>>(e2, mean, rstd, DMM*LSEQ);
  gn_apply<<<g1(BN*DMM*LSEQ), blk, 0, stream>>>(e2, mean, rstd, eg2, ebe2, e2, DMM, LSEQ, 1);

  nch_to_nhc<<<g1(BN*DMM*LSEQ), blk, 0, stream>>>(e2, sbuf);

  // -------- 6 Mamba layers --------
  for (int i = 0; i < NLL; ++i) {
    ln_rows<<<MROWS/8, blk, 0, stream>>>(sbuf, ln_g + i*DMM, ln_b + i*DMM,
                                         ulnh, nullptr, 0);
    // in-projection: (8192x128) x (512x128)^T -> xz
    wmma_gemm<4,128><<<dim3(MB, NXZ/64), blk, 0, stream>>>(
        ulnh, wih + (size_t)i*NXZ*DMM, nullptr, xz, MROWS, NXZ);

    dwconv_silu<<<g1(MROWS*DII), blk, 0, stream>>>(
        xz, c_W + (size_t)i*DII*DCC, c_b + (size_t)i*DII, xc, xch);

    // xp-projection: (8192x256) x (48x256)^T -> dbl
    wmma_gemm<3,256><<<dim3(MB, 1), blk, 0, stream>>>(
        xch, wxph + (size_t)i*NXP*DII, nullptr, dbl, MROWS, NXP);

    dt_softplus<<<g1(MROWS*DII), blk, 0, stream>>>(
        dbl, dt_W + (size_t)i*DII*DRR, dt_b + (size_t)i*DII, dtv);
    ssm_scan<<<(BN*DII)/16, blk, 0, stream>>>(
        dtv, xc, dbl, xz, A_log + (size_t)i*DII*DSZ, D_p + (size_t)i*DII, yh);

    // out-projection + residual (in-place on sbuf): (8192x256) x (128x256)^T
    wmma_gemm<4,256><<<dim3(MB, DMM/64), blk, 0, stream>>>(
        yh, woh + (size_t)i*DMM*DII, sbuf, sbuf, MROWS, DMM);
  }

  // -------- final LN, written transposed (B,DM,L) for decoder --------
  ln_rows<<<MROWS/8, blk, 0, stream>>>(sbuf, nf_g, nf_b, nullptr, decin, 1);

  // -------- decoder --------
  deconv1d<<<g1(BN*64*1024), blk, 0, stream>>>(decin, dW0, db0, dd0, DMM, 64, LSEQ, 1024);
  gn_stats<<<BN, blk, 0, stream>>>(dd0, mean, rstd, 64*1024);
  gn_apply<<<g1(BN*64*1024), blk, 0, stream>>>(dd0, mean, rstd, dg0, dbe0, dd0, 64, 1024, 1);

  deconv1d<<<g1(BN*32*2048), blk, 0, stream>>>(dd0, dW1, db1, dd1, 64, 32, 1024, 2048);
  gn_stats<<<BN, blk, 0, stream>>>(dd1, mean, rstd, 32*2048);
  gn_apply<<<g1(BN*32*2048), blk, 0, stream>>>(dd1, mean, rstd, dg1, dbe1, dd1, 32, 2048, 1);

  deconv1d<<<g1(NXT), blk, 0, stream>>>(dd1, dW2, db2, (float*)d_out, 32, CINN, 2048, TT);

  // -------- output tuple tail: (d, x, mask) --------
  tail_copy<<<g1(NXT), blk, 0, stream>>>(x, mask, (float*)d_out, NXT);
}
